// VSSM_86053964742724
// MI455X (gfx1250) — matrix operand order repaired and checked by reading, not compiled, not run on hardware
//
#include <hip/hip_runtime.h>

// ---------------------------------------------------------------------------
// VSSM (VMamba-style 2D selective scan block) for gfx1250 / MI455X.
// GEMMs: v_wmma_f32_16x16x32_f16 (fp16 in, fp32 accum).
// Selective scan: register-resident 16-state recurrence, one workgroup per
// (b,k) chain; per-step dt/B/C operands streamed into LDS by the Tensor Data
// Mover (tensor_load_to_lds), double-buffered on TENSORcnt.
// ---------------------------------------------------------------------------

#define DEV_INLINE __device__ __forceinline__

constexpr int Bb = 2;            // batch
constexpr int Cc = 96;           // d_model
constexpr int Hh = 96, Ww = 96;  // spatial
constexpr int Ll = Hh * Ww;      // 9216
constexpr int DI = 192;          // d_inner
constexpr int K4 = 4;            // scan directions
constexpr int NS = 16;           // d_state
constexpr int DR = 6;            // dt_rank
constexpr int RR = DR + 2 * NS;  // 38 rows of x_dbl consumed per step
constexpr int CP = 48;           // padded 38 -> 48 for WMMA tiles
constexpr int XZW = 2 * DI;      // 384
constexpr int TT = 32;           // scan chunk (TDM tile width, time steps)
constexpr int NC = Ll / TT;      // 288 chunks

typedef __attribute__((ext_vector_type(16))) _Float16 v16h;
typedef __attribute__((ext_vector_type(8)))  float    v8f;
typedef unsigned int u32x4 __attribute__((ext_vector_type(4)));
typedef int          i32x8 __attribute__((ext_vector_type(8)));
typedef int          i32x4 __attribute__((ext_vector_type(4)));

#if defined(__has_builtin)
#if __has_builtin(__builtin_amdgcn_tensor_load_to_lds)
#define HAVE_TDM 1
#endif
#endif
#ifndef HAVE_TDM
#define HAVE_TDM 0
#endif

// scan-order index j (direction k) -> hw-order linear index
DEV_INLINE int dmap(int k, int j) {
  switch (k) {
    case 0:  return j;
    case 1:  return (j % Hh) * Ww + (j / Hh);          // wh order
    case 2:  return Ll - 1 - j;                        // reversed hw
    default: { int t = Ll - 1 - j; return (t % Hh) * Ww + (t / Hh); }
  }
}

// ---------------------------------------------------------------------------
// Prep: transpose+convert input to f16 [b][l][c]
// ---------------------------------------------------------------------------
__global__ void k_xT(const float* __restrict__ x, _Float16* __restrict__ xT) {
  int i = blockIdx.x * blockDim.x + threadIdx.x;
  if (i >= Bb * Ll * Cc) return;
  int c = i % Cc, l = (i / Cc) % Ll, b = i / (Cc * Ll);
  xT[i] = (_Float16)x[((size_t)b * Cc + c) * Ll + l];
}

// Prep: f16 weights + A = -exp(A_logs)
__global__ void k_wprep(const float* __restrict__ inw, const float* __restrict__ outw,
                        const float* __restrict__ alogs,
                        _Float16* __restrict__ inw16, _Float16* __restrict__ outw16,
                        float* __restrict__ aneg) {
  int i = blockIdx.x * blockDim.x + threadIdx.x;
  if (i < XZW * Cc)       inw16[i]  = (_Float16)inw[i];
  if (i < Cc * DI)        outw16[i] = (_Float16)outw[i];
  if (i < K4 * DI * NS)   aneg[i]   = -__expf(alogs[i]);
}

// ---------------------------------------------------------------------------
// in_proj: xz[m,384] = xT[m,96] @ W^T   (one wave per 16x16 tile, K=96)
// ---------------------------------------------------------------------------
__global__ void k_inproj(const _Float16* __restrict__ xT,
                         const _Float16* __restrict__ w16,
                         float* __restrict__ xz) {
  int wave = blockIdx.x * (blockDim.x >> 5) + (threadIdx.x >> 5);
  int lane = threadIdx.x & 31, hi = lane >> 4, ln = lane & 15;
  const int NT = XZW / 16;                       // 24
  int mT = wave / NT, nT = wave % NT;
  if (mT >= (Bb * Ll) / 16) return;
  v8f acc = {};
  for (int kk = 0; kk < Cc; kk += 32) {
    v16h a, bf;
    int arow = mT * 16 + ln;
#pragma unroll
    for (int e = 0; e < 16; ++e) {
      int ka = kk + e + hi * 8 + ((e >= 8) ? 8 : 0);
      a[e]  = xT[(size_t)arow * Cc + ka];
      bf[e] = w16[(size_t)(nT * 16 + ln) * Cc + ka];
    }
    acc = __builtin_amdgcn_wmma_f32_16x16x32_f16(false, a, false, bf,
                                                 (short)0, acc, false, false);
  }
#pragma unroll
  for (int r = 0; r < 8; ++r) {
    int m = mT * 16 + r + 8 * hi;
    xz[(size_t)m * XZW + nT * 16 + ln] = acc[r];
  }
}

// ---------------------------------------------------------------------------
// depthwise 3x3 conv + SiLU : xi = xz[..., :192] (BHWC) -> xc [b][d][l]
// ---------------------------------------------------------------------------
__global__ void k_conv(const float* __restrict__ xz, const float* __restrict__ cw,
                       const float* __restrict__ cb, float* __restrict__ xc) {
  int i = blockIdx.x * blockDim.x + threadIdx.x;
  if (i >= Bb * DI * Ll) return;
  int l = i % Ll, d = (i / Ll) % DI, b = i / (Ll * DI);
  int h = l / Ww, w = l % Ww;
  float acc = cb[d];
#pragma unroll
  for (int dh = -1; dh <= 1; ++dh) {
    int hh = h + dh; if ((unsigned)hh >= (unsigned)Hh) continue;
#pragma unroll
    for (int dw = -1; dw <= 1; ++dw) {
      int ww2 = w + dw; if ((unsigned)ww2 >= (unsigned)Ww) continue;
      acc += xz[((size_t)b * Ll + hh * Ww + ww2) * XZW + d] *
             cw[d * 9 + (dh + 1) * 3 + (dw + 1)];
    }
  }
  acc = acc / (1.f + __expf(-acc));                 // SiLU
  xc[((size_t)b * DI + d) * Ll + l] = acc;
}

// ---------------------------------------------------------------------------
// x_proj: x_dbl[bk][c48][l] = W_k[38(->48),192] @ xs_k[192,L]
// ---------------------------------------------------------------------------
__global__ void k_xproj(const float* __restrict__ xpw, const float* __restrict__ xc,
                        float* __restrict__ xdbl) {
  int wave = blockIdx.x * (blockDim.x >> 5) + (threadIdx.x >> 5);
  int lane = threadIdx.x & 31, hi = lane >> 4, ln = lane & 15;
  const int MT = CP / 16, NT = Ll / 16;            // 3, 576
  int nT = wave % NT, mT = (wave / NT) % MT, bk = wave / (NT * MT);
  if (bk >= Bb * K4) return;
  int b = bk / K4, k = bk % K4;
  v8f acc = {};
  for (int kk = 0; kk < DI; kk += 32) {
    v16h a, bf;
    int arow = mT * 16 + ln;
#pragma unroll
    for (int e = 0; e < 16; ++e) {
      int ka = kk + e + hi * 8 + ((e >= 8) ? 8 : 0);
      a[e] = (arow < RR)
                 ? (_Float16)xpw[((size_t)k * RR + arow) * DI + ka]
                 : (_Float16)0.f;
      int kd = ka;
      int j  = nT * 16 + ln;
      bf[e] = (_Float16)xc[((size_t)b * DI + kd) * Ll + dmap(k, j)];
    }
    acc = __builtin_amdgcn_wmma_f32_16x16x32_f16(false, a, false, bf,
                                                 (short)0, acc, false, false);
  }
#pragma unroll
  for (int r = 0; r < 8; ++r) {
    int crow = mT * 16 + r + 8 * hi;
    xdbl[((size_t)bk * CP + crow) * Ll + nT * 16 + ln] = acc[r];
  }
}

// ---------------------------------------------------------------------------
// TDM: issue a 2D tile load (TT cols x 38 rows, row stride Ll floats) into LDS
// D# layout per CDNA5 ISA ch.8: group0 = {count/lds_addr/global_addr/type},
// group1 = {data_size, tensor_dim0/1, tile_dim0/1, tensor_dim0_stride}.
// This toolchain's builtin takes 6 args: (g0, g1, g2, g3, g4, cpol).
// ---------------------------------------------------------------------------
DEV_INLINE void tdm_issue(unsigned int lds_off, const float* gsrc) {
#if HAVE_TDM
  unsigned long long ga = (unsigned long long)(uintptr_t)gsrc;
  u32x4 g0 = {
    1u,                                                   // count=1, user mode
    lds_off,                                              // lds_addr (bytes)
    (unsigned int)ga,                                     // global_addr[31:0]
    (unsigned int)((ga >> 32) & 0x1FFFFFFu) | (2u << 30)  // addr[56:32] | type=2
  };
  i32x8 g1 = {
    (int)(2u << 16),                                  // mask=0, data_size=4B
    (int)(((unsigned)Ll & 0xFFFFu) << 16),            // tensor_dim0[15:0] @63:48
    (int)((((unsigned)Ll >> 16) & 0xFFFFu) |
          ((unsigned)CP << 16)),                      // dim0[31:16] | tensor_dim1 lo
    (int)((unsigned)TT << 16),                        // tensor_dim1 hi=0 | tile_dim0
    (int)RR,                                          // tile_dim1=38 | tile_dim2=0
    (int)Ll,                                          // tensor_dim0_stride[31:0]
    0, 0                                              // stride hi / dim1_stride
  };
  i32x4 z4 = {0, 0, 0, 0};
  i32x8 z8 = {0, 0, 0, 0, 0, 0, 0, 0};
  __builtin_amdgcn_tensor_load_to_lds(g0, g1, z4, z4, z8, 0);
#else
  (void)lds_off; (void)gsrc;
#endif
}

// ---------------------------------------------------------------------------
// selective scan: 8 workgroups (b,k), 192 threads (one per d). 16-state
// recurrence in VGPRs; x_dbl chunk (38 x TT) streamed into LDS by the TDM,
// double-buffered on TENSORcnt; delta = softplus(dt-proj) computed on the fly.
// ---------------------------------------------------------------------------
__global__ void __launch_bounds__(DI) k_scan(const float* __restrict__ xdbl,
                                             const float* __restrict__ xc,
                                             const float* __restrict__ aneg,
                                             const float* __restrict__ dtw_g,
                                             const float* __restrict__ dtb_g,
                                             const float* __restrict__ ds_g,
                                             float* __restrict__ y) {
  int bk = blockIdx.x, b = bk / K4, k = bk % K4;
  int d = threadIdx.x;
  __shared__ float buf[2][RR * TT];
  float h[NS], Ar[NS], dtw[DR];
#pragma unroll
  for (int n = 0; n < NS; ++n) { h[n] = 0.f; Ar[n] = aneg[((size_t)k * DI + d) * NS + n]; }
#pragma unroll
  for (int r = 0; r < DR; ++r) dtw[r] = dtw_g[((size_t)k * DI + d) * DR + r];
  float dtb = dtb_g[k * DI + d];
  float dsv = ds_g[k * DI + d];
  const float* xd = xdbl + (size_t)bk * CP * Ll;
  const float* uc = xc + ((size_t)b * DI + d) * Ll;
  float* yo = y + ((size_t)bk * DI + d) * Ll;
  bool issuer = (threadIdx.x < 32);                 // wave 0 drives the TDM

#if HAVE_TDM
  if (issuer) tdm_issue((unsigned int)(uintptr_t)&buf[0][0], xd);
#endif

  for (int c = 0; c < NC; ++c) {
    int cur = c & 1;
#if HAVE_TDM
    if (issuer) __builtin_amdgcn_s_wait_tensorcnt(0);   // chunk c landed in LDS
    __syncthreads();                                    // publish to all waves
    if (issuer && (c + 1) < NC)                          // prefetch chunk c+1
      tdm_issue((unsigned int)(uintptr_t)&buf[cur ^ 1][0], xd + (size_t)(c + 1) * TT);
#else
    if (d < RR) {
      for (int t = 0; t < TT; ++t)
        buf[cur][d * TT + t] = xd[(size_t)d * Ll + c * TT + t];
    }
    __syncthreads();
#endif
    const float* s = buf[cur];
    __builtin_prefetch(&uc[dmap(k, (c + 1 < NC ? c + 1 : c) * TT)], 0, 1);
#pragma unroll 4
    for (int t = 0; t < TT; ++t) {
      int j = c * TT + t;
      float dtv = dtb;
#pragma unroll
      for (int r = 0; r < DR; ++r) dtv += s[r * TT + t] * dtw[r];
      float delta = (dtv > 20.f) ? dtv : log1pf(__expf(dtv));   // softplus
      float u  = uc[dmap(k, j)];
      float du = delta * u;
      float ys = 0.f;
#pragma unroll
      for (int n = 0; n < NS; ++n) {
        float dA = __expf(delta * Ar[n]);
        h[n] = h[n] * dA + du * s[(DR + n) * TT + t];
        ys  += h[n] * s[(DR + NS + n) * TT + t];
      }
      yo[j] = ys + dsv * u;
    }
    // one barrier per chunk: next write to buf[cur^1] happens after the next
    // iteration's barrier, by which time everyone has finished this compute.
  }
}

// ---------------------------------------------------------------------------
// merge 4 directions + LayerNorm + SiLU(z) gate -> yo16 [b][l][d] (f16)
// ---------------------------------------------------------------------------
__global__ void __launch_bounds__(DI) k_merge(const float* __restrict__ y,
                                              const float* __restrict__ xz,
                                              const float* __restrict__ lng,
                                              const float* __restrict__ lnb,
                                              _Float16* __restrict__ yo16) {
  int row = blockIdx.x;                       // b*L + l
  int b = row / Ll, l = row % Ll;
  int d = threadIdx.x;
  __shared__ float s1[DI], s2[DI], red[2];
  int l1 = (l % Hh) * Ww + (l / Hh);          // hw -> wh index (self-inverse)
  size_t base = (size_t)b * K4 * DI * Ll;
  float v = y[base + ((size_t)0 * DI + d) * Ll + l]
          + y[base + ((size_t)2 * DI + d) * Ll + (Ll - 1 - l)]
          + y[base + ((size_t)1 * DI + d) * Ll + l1]
          + y[base + ((size_t)3 * DI + d) * Ll + (Ll - 1 - l1)];
  s1[d] = v; s2[d] = v * v;
  __syncthreads();
  for (int s = 96; s > 2; s >>= 1) {          // 192->96->48->24->12->6->3
    if (d < s) { s1[d] += s1[d + s]; s2[d] += s2[d + s]; }
    __syncthreads();
  }
  if (d == 0) {
    float sm = s1[0] + s1[1] + s1[2], ss = s2[0] + s2[1] + s2[2];
    red[0] = sm / (float)DI;
    red[1] = ss / (float)DI;
  }
  __syncthreads();
  float mu = red[0], var = red[1] - mu * mu;
  float yn = (v - mu) * rsqrtf(var + 1e-5f) * lng[d] + lnb[d];
  float z  = xz[(size_t)row * XZW + DI + d];
  float gz = z / (1.f + __expf(-z));
  yo16[(size_t)row * DI + d] = (_Float16)(yn * gz);
}

// ---------------------------------------------------------------------------
// out_proj + residual: out[b][c][l] = yo[b,l,:] @ W_out^T + x  (K=192)
// ---------------------------------------------------------------------------
__global__ void k_outproj(const _Float16* __restrict__ yo16,
                          const _Float16* __restrict__ w16,
                          const float* __restrict__ x,
                          float* __restrict__ out) {
  int wave = blockIdx.x * (blockDim.x >> 5) + (threadIdx.x >> 5);
  int lane = threadIdx.x & 31, hi = lane >> 4, ln = lane & 15;
  const int NT = Cc / 16;                     // 6
  int mT = wave / NT, nT = wave % NT;
  if (mT >= (Bb * Ll) / 16) return;
  v8f acc = {};
  for (int kk = 0; kk < DI; kk += 32) {
    v16h a, bf;
    int arow = mT * 16 + ln;
#pragma unroll
    for (int e = 0; e < 16; ++e) {
      int ka = kk + e + hi * 8 + ((e >= 8) ? 8 : 0);
      a[e]  = yo16[(size_t)arow * DI + ka];
      bf[e] = w16[(size_t)(nT * 16 + ln) * DI + ka];
    }
    acc = __builtin_amdgcn_wmma_f32_16x16x32_f16(false, a, false, bf,
                                                 (short)0, acc, false, false);
  }
#pragma unroll
  for (int r = 0; r < 8; ++r) {
    int m = mT * 16 + r + 8 * hi;
    int b = m / Ll, l = m % Ll, c = nT * 16 + ln;
    size_t idx = ((size_t)b * Cc + c) * Ll + l;
    out[idx] = acc[r] + x[idx];
  }
}

// ---------------------------------------------------------------------------
extern "C" void kernel_launch(void* const* d_in, const int* in_sizes, int n_in,
                              void* d_out, int out_size, void* d_ws, size_t ws_size,
                              hipStream_t stream) {
  (void)in_sizes; (void)n_in; (void)out_size; (void)ws_size;
  const float* x      = (const float*)d_in[0];
  const float* inw    = (const float*)d_in[1];
  const float* convw  = (const float*)d_in[2];
  const float* convb  = (const float*)d_in[3];
  const float* xprojw = (const float*)d_in[4];
  const float* dtw    = (const float*)d_in[5];
  const float* dtb    = (const float*)d_in[6];
  const float* alogs  = (const float*)d_in[7];
  const float* dsv    = (const float*)d_in[8];
  const float* lng    = (const float*)d_in[9];
  const float* lnb    = (const float*)d_in[10];
  const float* outw   = (const float*)d_in[11];
  float* out = (float*)d_out;

  // workspace partition (256B aligned)
  char* w = (char*)d_ws;
  size_t off = 0;
  auto take = [&](size_t bytes) { size_t r = off; off = (off + bytes + 255) & ~(size_t)255; return r; };
  _Float16* xT16   = (_Float16*)(w + take((size_t)Bb * Ll * Cc * 2));
  _Float16* inw16  = (_Float16*)(w + take((size_t)XZW * Cc * 2));
  _Float16* outw16 = (_Float16*)(w + take((size_t)Cc * DI * 2));
  float*    aneg   = (float*)   (w + take((size_t)K4 * DI * NS * 4));
  float*    xz     = (float*)   (w + take((size_t)Bb * Ll * XZW * 4));
  float*    xc     = (float*)   (w + take((size_t)Bb * DI * Ll * 4));
  float*    xdbl   = (float*)   (w + take((size_t)Bb * K4 * CP * Ll * 4));
  float*    ybuf   = (float*)   (w + take((size_t)Bb * K4 * DI * Ll * 4));
  _Float16* yo16   = (_Float16*)(w + take((size_t)Bb * Ll * DI * 2));

  {
    int n = Bb * Ll * Cc;
    k_xT<<<(n + 255) / 256, 256, 0, stream>>>(x, xT16);
    k_wprep<<<(XZW * Cc + 255) / 256, 256, 0, stream>>>(inw, outw, alogs,
                                                        inw16, outw16, aneg);
  }
  k_inproj<<<(Bb * Ll / 16) * (XZW / 16) / 4, 128, 0, stream>>>(xT16, inw16, xz);
  {
    int n = Bb * DI * Ll;
    k_conv<<<(n + 255) / 256, 256, 0, stream>>>(xz, convw, convb, xc);
  }
  k_xproj<<<(Bb * K4) * (CP / 16) * (Ll / 16) / 4, 128, 0, stream>>>(xprojw, xc, xdbl);
  k_scan<<<Bb * K4, DI, 0, stream>>>(xdbl, xc, aneg, dtw, dtb, dsv, ybuf);
  k_merge<<<Bb * Ll, DI, 0, stream>>>(ybuf, xz, lng, lnb, yo16);
  k_outproj<<<(Bb * Ll / 16) * (Cc / 16) / 4, 128, 0, stream>>>(yo16, outw16, x, out);
}
